// HAN_36687610643003
// MI455X (gfx1250) — compile-verified, matrix operand order checked by reading, not run on hardware
//
#include <hip/hip_runtime.h>
#include <math.h>

// ---------------------------------------------------------------------------
// HAN on MI455X (gfx1250): bf16 WMMA everywhere, async-LDS double buffering
// inside the GRU recurrence, ds_read_b128 A-fragments, WMMA attention proj.
// ---------------------------------------------------------------------------

#define B_   64
#define D_   30
#define S_   40
#define E_   200
#define H_   64
#define V_   50000
#define C_   5
#define NS_  (B_ * D_)        // 1920 sentences
#define ROWS_ (NS_ * S_)      // 76800 word rows
#define KXPAD_ 224            // E padded to multiple of 32
#define NCOL_ 192             // 2H gates + H candidate

typedef __attribute__((ext_vector_type(16))) __bf16 v16bf;
typedef __attribute__((ext_vector_type(8)))  float  v8f;

union AFrag {
  v16bf v;
  __bf16 e[16];
  uint4 q[2];
};

// ---- fragment loaders (CDNA5 ISA 16-bit A/B layouts, wave32) --------------
// A 16x32: lane m = lane&15, khalf = lane>>4; elem e -> K = (e&7)+8*khalf+(e>=8?16:0)
// => per lane: two contiguous 8-element (16B) runs -> two b128 loads.
__device__ inline AFrag load_afrag_global(const __bf16* __restrict__ row, int ktile, int lane) {
  AFrag f;
  const int khalf = lane >> 4;
  const __bf16* p = row + ktile * 32 + 8 * khalf;
  f.q[0] = *(const uint4*)(p);
  f.q[1] = *(const uint4*)(p + 16);
  return f;
}

// Same pattern out of LDS (bf16 mirror, 16B-aligned row stride) -> 2x ds_read_b128
__device__ inline AFrag load_afrag_lds(const __bf16* row0, int stride, int ktile, int lane) {
  AFrag f;
  const int m = lane & 15, khalf = lane >> 4;
  const __bf16* p = row0 + m * stride + ktile * 32 + 8 * khalf;
  f.q[0] = *(const uint4*)(p);
  f.q[1] = *(const uint4*)(p + 16);
  return f;
}

// B 32x16: lane n = lane&15, khalf = lane>>4; same K mapping as A (gathered once).
__device__ inline AFrag load_bfrag(const __bf16* __restrict__ W, int ldw, int ntile, int ktile, int lane) {
  AFrag f;
  const int khalf = lane >> 4;
  const int ncol = ntile * 16 + (lane & 15);
  const int kbase = ktile * 32;
#pragma unroll
  for (int e = 0; e < 16; ++e) {
    int kk = kbase + (e & 7) + 8 * khalf + ((e >= 8) ? 16 : 0);
    f.e[e] = W[(size_t)kk * ldw + ncol];
  }
  return f;
}

// ---------------------------------------------------------------------------
__global__ void lens_kernel(const int* __restrict__ ids, int* sent_lens, int* doc_lens) {
  int idx = blockIdx.x * blockDim.x + threadIdx.x;
  if (idx < NS_) {
    const int* p = ids + (size_t)idx * S_;
    int c = 0;
    for (int s = 0; s < S_; ++s) c += (p[s] != 0);
    sent_lens[idx] = c;
  } else if (idx < NS_ + B_) {
    int b = idx - NS_;
    int c = 0;
    for (int d = 0; d < D_; ++d) {
      const int* p = ids + (size_t)(b * D_ + d) * S_;
      long long sum = 0;
      for (int s = 0; s < S_; ++s) sum += p[s];
      c += (sum != 0);
    }
    doc_lens[b] = c;
  }
}

__global__ void embed_kernel(const int* __restrict__ ids, const float* __restrict__ emb,
                             __bf16* __restrict__ Xb) {
  size_t idx = (size_t)blockIdx.x * blockDim.x + threadIdx.x;
  size_t total = (size_t)ROWS_ * KXPAD_;
  if (idx >= total) return;
  int row = (int)(idx / KXPAD_);
  int k   = (int)(idx % KXPAD_);
  float v = 0.f;
  if (k < E_) v = emb[(size_t)ids[row] * E_ + k];
  Xb[idx] = (__bf16)v;
}

__global__ void pack_weights_kernel(const float* __restrict__ Wg, const float* __restrict__ Wc,
                                    __bf16* __restrict__ Wx, __bf16* __restrict__ Wh,
                                    int din, int Kpad) {
  int idx = blockIdx.x * blockDim.x + threadIdx.x;
  int total = (Kpad + H_) * NCOL_;
  if (idx >= total) return;
  int row = idx / NCOL_;
  int col = idx % NCOL_;
  if (row < Kpad) {
    float v = 0.f;
    if (row < din)
      v = (col < 2 * H_) ? Wg[(size_t)row * (2 * H_) + col]
                         : Wc[(size_t)row * H_ + (col - 2 * H_)];
    Wx[(size_t)row * NCOL_ + col] = (__bf16)v;
  } else {
    int r = row - Kpad;
    float v = (col < 2 * H_) ? Wg[(size_t)(din + r) * (2 * H_) + col]
                             : Wc[(size_t)(din + r) * H_ + (col - 2 * H_)];
    Wh[(size_t)r * NCOL_ + col] = (__bf16)v;
  }
}

__global__ void f32_to_bf16_kernel(const float* __restrict__ in, __bf16* __restrict__ out, int n) {
  int idx = blockIdx.x * blockDim.x + threadIdx.x;
  if (idx < n) out[idx] = (__bf16)in[idx];
}

// ---------------------------------------------------------------------------
// G[nrow,192] = X[nrow,K] @ W[K,192], bf16 in, f32 accumulate, bf16 out.
template <int KT>
__global__ void gemm_bf16_kernel(const __bf16* __restrict__ X, const __bf16* __restrict__ W,
                                 __bf16* __restrict__ G, int nrow) {
  const int lane = threadIdx.x & 31;
  const int wave = threadIdx.x >> 5;
  const int ntile = blockIdx.y * 4 + wave;          // 0..11
  const int ncol = ntile * 16 + (lane & 15);
  const int khalf = lane >> 4;
  const int K = KT * 32;

  AFrag bfrag[KT];
#pragma unroll
  for (int k = 0; k < KT; ++k) bfrag[k] = load_bfrag(W, NCOL_, ntile, k, lane);

  const int mtiles = nrow >> 4;
  for (int mt = blockIdx.x; mt < mtiles; mt += gridDim.x) {
    const __bf16* xrow = X + (size_t)(mt * 16 + (lane & 15)) * K;
    v8f acc = {};
#pragma unroll
    for (int k = 0; k < KT; ++k) {
      AFrag a = load_afrag_global(xrow, k, lane);
      acc = __builtin_amdgcn_wmma_f32_16x16x32_bf16(false, a.v, false, bfrag[k].v,
                                                    (short)0, acc, false, false);
    }
#pragma unroll
    for (int r = 0; r < 8; ++r) {
      int m = mt * 16 + r + 8 * khalf;
      G[(size_t)m * NCOL_ + ncol] = (__bf16)acc[r];
    }
  }
}

// ---------------------------------------------------------------------------
// GRU recurrence, 16 sequences / block (4 waves), one direction per blockIdx.y.
// G tile for step t+1 is async-copied into an LDS double buffer while step t
// computes (ASYNCcnt path); h / r*h kept as bf16 LDS mirrors so A-fragments
// are two ds_read_b128 each.
template <int T>
__global__ void gru_kernel(const __bf16* __restrict__ Wh_fw, const __bf16* __restrict__ Wh_bw,
                           const __bf16* __restrict__ G_fw, const __bf16* __restrict__ G_bw,
                           const float* __restrict__ bg_fw, const float* __restrict__ bc_fw,
                           const float* __restrict__ bg_bw, const float* __restrict__ bc_bw,
                           const int* __restrict__ lens, __bf16* __restrict__ enc) {
  __shared__ float  h_lds[16][64];        // f32 master state
  __shared__ __bf16 hb_lds[16][80];       // bf16 mirror, 160B stride (16B aligned)
  __shared__ __bf16 rhb_lds[16][80];
  __shared__ float  g_lds[16][128];
  __shared__ float  c_lds[16][64];
  __shared__ __bf16 gbuf[2][16 * NCOL_];  // async double buffer: 2 x 6144 B
  __shared__ int    len_lds[16];

  const int tid = threadIdx.x;
  const int lane = tid & 31;
  const int wave = tid >> 5;
  const int khalf = lane >> 4;
  const int dir = blockIdx.y;             // uniform
  const int seqbase = blockIdx.x * 16;

  const __bf16* Wh = dir ? Wh_bw : Wh_fw;
  const __bf16* G  = dir ? G_bw  : G_fw;
  const float*  bg = dir ? bg_bw : bg_fw;
  const float*  bc = dir ? bc_bw : bc_fw;

  if (tid < 16) len_lds[tid] = lens[seqbase + tid];
  {
    int m = tid >> 3, kb = (tid & 7) * 8;
#pragma unroll
    for (int i = 0; i < 8; ++i) { h_lds[m][kb + i] = 0.f; hb_lds[m][kb + i] = (__bf16)0.f; }
  }
  __syncthreads();

  // Wh B-fragments resident in registers: gate ntiles {2w,2w+1}, cand ntile 8+w
  AFrag Bg[2][2], Bc[2];
#pragma unroll
  for (int g = 0; g < 2; ++g)
#pragma unroll
    for (int k = 0; k < 2; ++k) Bg[g][k] = load_bfrag(Wh, NCOL_, 2 * wave + g, k, lane);
#pragma unroll
  for (int k = 0; k < 2; ++k) Bc[k] = load_bfrag(Wh, NCOL_, 8 + wave, k, lane);

  // async prefetch of one step's 16x192 bf16 G tile (384 x 16B chunks, 3/thread)
  auto prefetch = [&](int tstep, int buf) {
    unsigned lds_base = (unsigned)(size_t)(&gbuf[buf][0]);
#pragma unroll
    for (int i = 0; i < 3; ++i) {
      int c = tid + i * 128;                 // 0..383
      int m = c / 24;
      int off = (c % 24) * 16;               // bytes within 384B row
      int len = len_lds[m];
      int tt = dir ? ((tstep < len) ? len - 1 - tstep : tstep) : tstep;
      unsigned voff = (unsigned)((((size_t)(seqbase + m) * T + tt) * NCOL_) * 2 + off);
      unsigned ldsa = lds_base + (unsigned)(m * (NCOL_ * 2) + off);
      asm volatile("global_load_async_to_lds_b128 %0, %1, %2"
                   :: "v"(ldsa), "v"(voff), "s"(G) : "memory");
    }
  };

  prefetch(0, 0);

  for (int t = 0; t < T; ++t) {
    const __bf16* gcur = &gbuf[t & 1][0];
    if (t + 1 < T) {
      prefetch(t + 1, (t + 1) & 1);
      asm volatile("s_wait_asynccnt 0x3" ::: "memory");  // oldest 3 (this step) done
    } else {
      asm volatile("s_wait_asynccnt 0x0" ::: "memory");
    }
    __syncthreads();  // async LDS writes visible to all waves

    // ---- phase 1: gates = sigmoid(h@Whg + Gx + bg) ----
    {
      AFrag A0 = load_afrag_lds(&hb_lds[0][0], 80, 0, lane);
      AFrag A1 = load_afrag_lds(&hb_lds[0][0], 80, 1, lane);
#pragma unroll
      for (int g = 0; g < 2; ++g) {
        v8f acc = {};
        acc = __builtin_amdgcn_wmma_f32_16x16x32_bf16(false, A0.v, false, Bg[g][0].v,
                                                      (short)0, acc, false, false);
        acc = __builtin_amdgcn_wmma_f32_16x16x32_bf16(false, A1.v, false, Bg[g][1].v,
                                                      (short)0, acc, false, false);
        int col = (2 * wave + g) * 16 + (lane & 15);
#pragma unroll
        for (int r = 0; r < 8; ++r) {
          int m = r + 8 * khalf;
          float pre = acc[r] + (float)gcur[m * NCOL_ + col] + bg[col];
          g_lds[m][col] = 1.0f / (1.0f + __expf(-pre));
        }
      }
    }
    __syncthreads();

    // ---- r*h (bf16, A-layout-friendly) ----
    {
      int m = tid >> 3, kb = (tid & 7) * 8;
#pragma unroll
      for (int i = 0; i < 8; ++i) {
        int k = kb + i;
        rhb_lds[m][k] = (__bf16)(g_lds[m][k] * h_lds[m][k]);
      }
    }
    __syncthreads();

    // ---- phase 2: c = tanh((r*h)@Whc + Gx_c + bc) ----
    {
      AFrag A0 = load_afrag_lds(&rhb_lds[0][0], 80, 0, lane);
      AFrag A1 = load_afrag_lds(&rhb_lds[0][0], 80, 1, lane);
      v8f acc = {};
      acc = __builtin_amdgcn_wmma_f32_16x16x32_bf16(false, A0.v, false, Bc[0].v,
                                                    (short)0, acc, false, false);
      acc = __builtin_amdgcn_wmma_f32_16x16x32_bf16(false, A1.v, false, Bc[1].v,
                                                    (short)0, acc, false, false);
      int ccol = wave * 16 + (lane & 15);
#pragma unroll
      for (int r = 0; r < 8; ++r) {
        int m = r + 8 * khalf;
        float pre = acc[r] + (float)gcur[m * NCOL_ + 128 + ccol] + bc[ccol];
        c_lds[m][ccol] = tanhf(pre);
      }
    }
    __syncthreads();

    // ---- update + masked output (bf16 enc) ----
    {
      int m = tid >> 3, kb = (tid & 7) * 8;
      int len = len_lds[m];
      int valid = (t < len);
      int p = dir ? (valid ? len - 1 - t : t) : t;
      size_t erow = ((size_t)(seqbase + m) * T + p) * 128 + dir * 64;
#pragma unroll
      for (int i = 0; i < 8; ++i) {
        int k = kb + i;
        float h = h_lds[m][k];
        float u = g_lds[m][64 + k];
        float hn = u * h + (1.0f - u) * c_lds[m][k];
        float hv = valid ? hn : h;
        h_lds[m][k] = hv;
        hb_lds[m][k] = (__bf16)hv;
        enc[erow + k] = (__bf16)(valid ? hn : 0.0f);
      }
    }
    __syncthreads();
  }
}

// ---------------------------------------------------------------------------
// attention: proj = tanh(enc@W + b) via WMMA (K=128 -> 4 k-tiles, wave owns a
// 16-col n-tile); then softmax(proj.ctx) and weighted sum on VALU.
template <int T>
__global__ void attention_kernel(const __bf16* __restrict__ encb, const __bf16* __restrict__ Wb,
                                 const float* __restrict__ b, const float* __restrict__ ctx,
                                 float* __restrict__ att) {
  constexpr int MT = (T + 15) / 16;
  __shared__ float proj[MT * 16][64];
  __shared__ float wts[MT * 16];
  const int tid = threadIdx.x;
  const int lane = tid & 31;
  const int wave = tid >> 5;
  const int khalf = lane >> 4;
  const int n = blockIdx.x;
  const int ncol = wave * 16 + (lane & 15);

  AFrag Bf[4];
#pragma unroll
  for (int k = 0; k < 4; ++k) Bf[k] = load_bfrag(Wb, 64, wave, k, lane);
  const float bias = b[ncol];

#pragma unroll
  for (int mt = 0; mt < MT; ++mt) {
    const __bf16* arow = encb + ((size_t)n * T + mt * 16 + (lane & 15)) * 128;
    v8f acc = {};
#pragma unroll
    for (int k = 0; k < 4; ++k) {
      AFrag a = load_afrag_global(arow, k, lane);
      acc = __builtin_amdgcn_wmma_f32_16x16x32_bf16(false, a.v, false, Bf[k].v,
                                                    (short)0, acc, false, false);
    }
#pragma unroll
    for (int r = 0; r < 8; ++r) {
      int m = mt * 16 + r + 8 * khalf;
      if (m < T) proj[m][ncol] = tanhf(acc[r] + bias);   // masked store (post-WMMA)
    }
  }
  __syncthreads();

  if (tid < T) {
    float s = 0.f;
    for (int k = 0; k < 64; ++k) s += proj[tid][k] * ctx[k];
    wts[tid] = s;
  }
  __syncthreads();
  if (tid == 0) {
    float mx = wts[0];
    for (int t = 1; t < T; ++t) mx = fmaxf(mx, wts[t]);
    float z = 0.f;
    for (int t = 0; t < T; ++t) { wts[t] = __expf(wts[t] - mx); z += wts[t]; }
    float inv = 1.0f / z;
    for (int t = 0; t < T; ++t) wts[t] *= inv;
  }
  __syncthreads();
  if (tid < 64) {
    float a = 0.f;
    for (int t = 0; t < T; ++t) a += proj[t][tid] * wts[t];
    att[(size_t)n * 64 + tid] = a;
  }
}

__global__ void classifier_kernel(const float* __restrict__ att, const float* __restrict__ Wc,
                                  const float* __restrict__ bc, float* __restrict__ out) {
  int bidx = blockIdx.x;
  int c = threadIdx.x;
  if (c < C_) {
    float s = bc[c];
    for (int k = 0; k < H_; ++k) s += att[(size_t)bidx * H_ + k] * Wc[k * C_ + c];
    out[bidx * C_ + c] = s;
  }
}

// ---------------------------------------------------------------------------
static inline size_t align256(size_t x) { return (x + 255) & ~(size_t)255; }

extern "C" void kernel_launch(void* const* d_in, const int* in_sizes, int n_in,
                              void* d_out, int out_size, void* d_ws, size_t ws_size,
                              hipStream_t stream) {
  const int*   ids  = (const int*)d_in[0];
  const float* emb  = (const float*)d_in[1];
  const float* Wg[4] = { (const float*)d_in[2], (const float*)d_in[6],
                         (const float*)d_in[10], (const float*)d_in[14] };
  const float* bg[4] = { (const float*)d_in[3], (const float*)d_in[7],
                         (const float*)d_in[11], (const float*)d_in[15] };
  const float* Wc[4] = { (const float*)d_in[4], (const float*)d_in[8],
                         (const float*)d_in[12], (const float*)d_in[16] };
  const float* bc[4] = { (const float*)d_in[5], (const float*)d_in[9],
                         (const float*)d_in[13], (const float*)d_in[17] };
  const float* satt_W = (const float*)d_in[18];
  const float* satt_b = (const float*)d_in[19];
  const float* satt_c = (const float*)d_in[20];
  const float* datt_W = (const float*)d_in[21];
  const float* datt_b = (const float*)d_in[22];
  const float* datt_c = (const float*)d_in[23];
  const float* cls_W  = (const float*)d_in[24];
  const float* cls_b  = (const float*)d_in[25];
  float* out = (float*)d_out;

  // ---- workspace carve-out ----
  char* ws = (char*)d_ws;
  size_t cur = 0;
  auto alloc = [&](size_t bytes) { char* p = ws + cur; cur += align256(bytes); return p; };

  int*    sent_lens = (int*)alloc(NS_ * sizeof(int));
  int*    doc_lens  = (int*)alloc(B_ * sizeof(int));
  __bf16* Xb        = (__bf16*)alloc((size_t)ROWS_ * KXPAD_ * 2);
  __bf16* Wx_p[4], *Wh_p[4];
  for (int i = 0; i < 4; ++i) {
    Wx_p[i] = (__bf16*)alloc((size_t)KXPAD_ * NCOL_ * 2);
    Wh_p[i] = (__bf16*)alloc((size_t)H_ * NCOL_ * 2);
  }
  __bf16* G_sfw = (__bf16*)alloc((size_t)ROWS_ * NCOL_ * 2);
  __bf16* G_sbw = (__bf16*)alloc((size_t)ROWS_ * NCOL_ * 2);
  __bf16* enc_s = (__bf16*)alloc((size_t)NS_ * S_ * 128 * 2 + 8192);   // slack for tail M-tile reads
  float*  att_s = (float*)alloc((size_t)NS_ * H_ * sizeof(float));
  __bf16* Xd    = (__bf16*)alloc((size_t)NS_ * H_ * 2);
  __bf16* G_dfw = (__bf16*)alloc((size_t)NS_ * NCOL_ * 2);
  __bf16* G_dbw = (__bf16*)alloc((size_t)NS_ * NCOL_ * 2);
  __bf16* enc_d = (__bf16*)alloc((size_t)B_ * D_ * 128 * 2 + 8192);
  float*  att_d = (float*)alloc((size_t)B_ * H_ * sizeof(float));
  __bf16* sattWb = (__bf16*)alloc((size_t)128 * 64 * 2);
  __bf16* dattWb = (__bf16*)alloc((size_t)128 * 64 * 2);
  (void)ws_size; (void)in_sizes; (void)n_in; (void)out_size;

  // 1. sequence lengths
  lens_kernel<<<(NS_ + B_ + 255) / 256, 256, 0, stream>>>(ids, sent_lens, doc_lens);

  // 2. embedding gather -> bf16 padded
  {
    size_t total = (size_t)ROWS_ * KXPAD_;
    embed_kernel<<<(unsigned)((total + 255) / 256), 256, 0, stream>>>(ids, emb, Xb);
  }

  // 3. pack weights (sentence dirs: din=E; doc dirs: din=H) + attention W -> bf16
  for (int i = 0; i < 4; ++i) {
    int din  = (i < 2) ? E_ : H_;
    int kpad = (i < 2) ? KXPAD_ : H_;
    int total = (kpad + H_) * NCOL_;
    pack_weights_kernel<<<(total + 255) / 256, 256, 0, stream>>>(Wg[i], Wc[i], Wx_p[i], Wh_p[i], din, kpad);
  }
  f32_to_bf16_kernel<<<(128 * 64 + 255) / 256, 256, 0, stream>>>(satt_W, sattWb, 128 * 64);
  f32_to_bf16_kernel<<<(128 * 64 + 255) / 256, 256, 0, stream>>>(datt_W, dattWb, 128 * 64);

  // 4. big input GEMMs (hoisted x@Wx; memory-bound; bf16 WMMA)
  gemm_bf16_kernel<7><<<dim3(300, 3), 128, 0, stream>>>(Xb, Wx_p[0], G_sfw, ROWS_);
  gemm_bf16_kernel<7><<<dim3(300, 3), 128, 0, stream>>>(Xb, Wx_p[1], G_sbw, ROWS_);

  // 5. sentence-level bi-GRU (WMMA recurrence + async G prefetch)
  gru_kernel<S_><<<dim3(NS_ / 16, 2), 128, 0, stream>>>(
      Wh_p[0], Wh_p[1], G_sfw, G_sbw, bg[0], bc[0], bg[1], bc[1], sent_lens, enc_s);

  // 6. sentence attention (WMMA projection)
  attention_kernel<S_><<<NS_, 128, 0, stream>>>(enc_s, sattWb, satt_b, satt_c, att_s);

  // 7. doc-level input GEMMs
  f32_to_bf16_kernel<<<(NS_ * H_ + 255) / 256, 256, 0, stream>>>(att_s, Xd, NS_ * H_);
  gemm_bf16_kernel<2><<<dim3(120, 3), 128, 0, stream>>>(Xd, Wx_p[2], G_dfw, NS_);
  gemm_bf16_kernel<2><<<dim3(120, 3), 128, 0, stream>>>(Xd, Wx_p[3], G_dbw, NS_);

  // 8. doc-level bi-GRU
  gru_kernel<D_><<<dim3(B_ / 16, 2), 128, 0, stream>>>(
      Wh_p[2], Wh_p[3], G_dfw, G_dbw, bg[2], bc[2], bg[3], bc[3], doc_lens, enc_d);

  // 9. doc attention + classifier
  attention_kernel<D_><<<B_, 128, 0, stream>>>(enc_d, dattWb, datt_b, datt_c, att_d);
  classifier_kernel<<<B_, 32, 0, stream>>>(att_d, cls_W, cls_b, out);
}
